// LSTM_52922587021316
// MI455X (gfx1250) — matrix operand order repaired and checked by reading, compiled, not run on hardware
//
#include <hip/hip_runtime.h>
#include <math.h>

// CDNA5 (gfx1250) WMMA vector types
typedef __attribute__((ext_vector_type(16))) __bf16 v16bf;
typedef __attribute__((ext_vector_type(8)))  float  v8f;

#define HID 32
#define SEQT 7
#define GATES 128          // 4*HID
#define BATCH 262144
#define WPB 8              // waves per block
#define ROWS 16            // batch rows per wave (one WMMA M-tile)

// ---- branch-free transcendentals (TRANS32 pipe, co-executes with WMMA) ----
#if __has_builtin(__builtin_amdgcn_exp2f)
#define EXP2F(x) __builtin_amdgcn_exp2f(x)
#else
#define EXP2F(x) exp2f(x)
#endif
#if __has_builtin(__builtin_amdgcn_rcpf)
#define RCPF(x) __builtin_amdgcn_rcpf(x)
#else
#define RCPF(x) (1.0f / (x))
#endif

__device__ __forceinline__ float fast_sig(float x) {
  // rcp(1 + 2^(-x*log2 e)) : saturates correctly at +-inf, no branches
  return RCPF(1.0f + EXP2F(-1.4426950408889634f * x));
}
__device__ __forceinline__ float fast_tanh(float x) {
#if __has_builtin(__builtin_amdgcn_tanhf)
  return __builtin_amdgcn_tanhf(x);         // v_tanh_f32
#else
  // tanh(x) = 1 - 2/(exp(2x)+1), branch-free, saturates at +-1
  return 1.0f - 2.0f * RCPF(1.0f + EXP2F(2.8853900817779268f * x));
#endif
}

// wave-private LDS ordering: DS ops of a wave are in-order (ISA 7.3); this
// forces compile-order + completion without a block-wide barrier.
#define LDS_MEMBAR() asm volatile("s_wait_dscnt 0" ::: "memory")

// Build the 16x32 bf16 A-fragment (ISA 7.12.2, 16-bit A 16x32 layout):
//   lanes 0-15 / 16-31 both hold rows M=0..15 (row = lane&15)
//   VGPR j (elements 2j,2j+1): K = (j>=4)*16 + (lane>>4)*8 + (j&3)*2 + {0,1}
__device__ __forceinline__ v16bf load_afrag(const float* hbuf,
                                            int lanelo, int lanehi) {
  v16bf a;
#pragma unroll
  for (int j = 0; j < 8; ++j) {
    const int kb = ((j >= 4) ? 16 : 0) + lanehi * 8 + (j & 3) * 2;
    a[2 * j]     = (__bf16)hbuf[lanelo * HID + kb];
    a[2 * j + 1] = (__bf16)hbuf[lanelo * HID + kb + 1];
  }
  return a;
}

__global__ __launch_bounds__(256)
void lstm2_fc_wmma_kernel(const float* __restrict__ inp,
                          const float* __restrict__ Wih0, const float* __restrict__ Whh0,
                          const float* __restrict__ bih0, const float* __restrict__ bhh0,
                          const float* __restrict__ Wih1, const float* __restrict__ Whh1,
                          const float* __restrict__ bih1, const float* __restrict__ bhh1,
                          const float* __restrict__ fcw,  const float* __restrict__ fcb,
                          const float* __restrict__ fc1w, const float* __restrict__ fc1b,
                          float* __restrict__ out)
{
  __shared__ float sB0[GATES], sB1[GATES], sWih0[GATES];
  __shared__ float sFcw[HID];
  __shared__ float sFc1w[SEQT];
  __shared__ float sScal[2];
  __shared__ float sH0[WPB][ROWS * HID];
  __shared__ float sH1[WPB][ROWS * HID];
  __shared__ float sX[WPB][ROWS * 8];     // 16 rows x 7 timesteps, stride 8

  const int tid    = threadIdx.x;
  const int lane   = tid & 31;
  const int wave   = tid >> 5;
  const int lanelo = lane & 15;   // N col within tile; M row for A frags
  const int lanehi = lane >> 4;   // K-half / M-half selector
  const long long base = ((long long)blockIdx.x * WPB + wave) * ROWS;

  // ---- stage small parameters (cross-wave) ----
  if (tid < GATES) {
    sB0[tid]   = bih0[tid] + bhh0[tid];
    sB1[tid]   = bih1[tid] + bhh1[tid];
    sWih0[tid] = Wih0[tid];
  }
  if (tid < HID)  sFcw[tid]  = fcw[tid];
  if (tid < SEQT) sFc1w[tid] = fc1w[tid];
  if (tid == 0) { sScal[0] = fcb[0]; sScal[1] = fc1b[0]; }

  float* h0buf = sH0[wave];
  float* h1buf = sH1[wave];
  float* xbuf  = sX[wave];

  // wave-private staging: zero h state, coalesced prefetch of the x block
  for (int i = lane; i < ROWS * HID; i += 32) { h0buf[i] = 0.0f; h1buf[i] = 0.0f; }
  for (int i = lane; i < ROWS * SEQT; i += 32) {
    const int row = i / SEQT, tt = i - row * SEQT;
    xbuf[row * 8 + tt] = inp[(base + row) * SEQT + tt];
  }
  __syncthreads();   // the only block barrier: params + own staging visible

  // ---- pack persistent B-fragments (ISA 7.12.2, 16-bit B KxN layout):
  //   lane = N within tile (lanes 0-15: K=0..7,16..23, lanes 16-31: K=8..15,24..31)
  //   element e <-> K = lanehi*8 + e + (e&8) ; B[k][n] = W[n_global][k]
  v16bf fhh0[8], fih1[8], fhh1[8];
#pragma unroll
  for (int nt = 0; nt < 8; ++nt) {
    const int ng = nt * 16 + lanelo;
    const int kb = lanehi * 8;
#pragma unroll
    for (int e = 0; e < 16; ++e) {
      fhh0[nt][e] = (__bf16)Whh0[ng * HID + kb + e + (e & 8)];
      fih1[nt][e] = (__bf16)Wih1[ng * HID + kb + e + (e & 8)];
      fhh1[nt][e] = (__bf16)Whh1[ng * HID + kb + e + (e & 8)];
    }
  }

  // cell states (C/D layout: elem r -> row M = r + 8*lanehi, col = j*16+lanelo)
  v8f c0[2], c1[2];
#pragma unroll
  for (int j = 0; j < 2; ++j)
#pragma unroll
    for (int r = 0; r < 8; ++r) { c0[j][r] = 0.0f; c1[j][r] = 0.0f; }

  float accp[8];                 // per-lane partial of fc1-weighted fc dot
#pragma unroll
  for (int r = 0; r < 8; ++r) accp[r] = 0.0f;

#pragma unroll
  for (int t = 0; t < SEQT; ++t) {
    // A-fragments of current hidden states (reads before any overwrite)
    v16bf a0 = load_afrag(h0buf, lanelo, lanehi);
    v16bf a1 = load_afrag(h1buf, lanelo, lanehi);
    LDS_MEMBAR();

    // x_t for this lane's 8 rows (broadcast reads from wave-private LDS)
    float xr[8];
#pragma unroll
    for (int r = 0; r < 8; ++r) xr[r] = xbuf[(lanehi * 8 + r) * 8 + t];

    // ================= layer 0 =================
    float hnew0[2][8];
#pragma unroll
    for (int j = 0; j < 2; ++j) {           // hidden col block j*16..j*16+15
      v8f acc[4];
#pragma unroll
      for (int q = 0; q < 4; ++q) {         // q: 0=i 1=f 2=g 3=o
        const int nt = j + 2 * q;
        const int ng = nt * 16 + lanelo;
        const float b = sB0[ng], w = sWih0[ng];
        v8f cfr;                            // C = bias + rank-1 input term (f32)
#pragma unroll
        for (int r = 0; r < 8; ++r) cfr[r] = b + xr[r] * w;
        acc[q] = __builtin_amdgcn_wmma_f32_16x16x32_bf16(
            false, a0, false, fhh0[nt], (short)0, cfr, false, false);
      }
#pragma unroll
      for (int r = 0; r < 8; ++r) {
        const float ig = fast_sig(acc[0][r]);
        const float fg = fast_sig(acc[1][r]);
        const float gg = fast_tanh(acc[2][r]);
        const float og = fast_sig(acc[3][r]);
        const float c  = fg * c0[j][r] + ig * gg;
        c0[j][r] = c;
        hnew0[j][r] = og * fast_tanh(c);
      }
    }
#pragma unroll
    for (int j = 0; j < 2; ++j)
#pragma unroll
      for (int r = 0; r < 8; ++r)
        h0buf[(lanehi * 8 + r) * HID + j * 16 + lanelo] = hnew0[j][r];
    LDS_MEMBAR();

    v16bf ax = load_afrag(h0buf, lanelo, lanehi);   // layer-1 input fragment

    // ================= layer 1 =================
    float hnew1[2][8];
#pragma unroll
    for (int j = 0; j < 2; ++j) {
      v8f acc[4];
#pragma unroll
      for (int q = 0; q < 4; ++q) {
        const int nt = j + 2 * q;
        const int ng = nt * 16 + lanelo;
        const float b = sB1[ng];
        v8f cfr;
#pragma unroll
        for (int r = 0; r < 8; ++r) cfr[r] = b;
        v8f tmp = __builtin_amdgcn_wmma_f32_16x16x32_bf16(
            false, ax, false, fih1[nt], (short)0, cfr, false, false);
        acc[q] = __builtin_amdgcn_wmma_f32_16x16x32_bf16(
            false, a1, false, fhh1[nt], (short)0, tmp, false, false);
      }
#pragma unroll
      for (int r = 0; r < 8; ++r) {
        const float ig = fast_sig(acc[0][r]);
        const float fg = fast_sig(acc[1][r]);
        const float gg = fast_tanh(acc[2][r]);
        const float og = fast_sig(acc[3][r]);
        const float c  = fg * c1[j][r] + ig * gg;
        c1[j][r] = c;
        hnew1[j][r] = og * fast_tanh(c);
      }
    }
#pragma unroll
    for (int j = 0; j < 2; ++j)
#pragma unroll
      for (int r = 0; r < 8; ++r)
        h1buf[(lanehi * 8 + r) * HID + j * 16 + lanelo] = hnew1[j][r];

    // fc partials: defer cross-lane reduction to the end (it's linear)
    const float wt = sFc1w[t];
#pragma unroll
    for (int r = 0; r < 8; ++r)
      accp[r] += wt * (hnew1[0][r] * sFcw[lanelo] + hnew1[1][r] * sFcw[16 + lanelo]);

    LDS_MEMBAR();   // order h1buf store before next iteration's a1 read
  }

  // one-shot reduction across the 16-lane half groups (wave32)
  float sumw = 0.0f;
#pragma unroll
  for (int t = 0; t < SEQT; ++t) sumw += sFc1w[t];

#pragma unroll
  for (int r = 0; r < 8; ++r) {
    float p = accp[r];
    p += __shfl_xor(p, 1, 32);
    p += __shfl_xor(p, 2, 32);
    p += __shfl_xor(p, 4, 32);
    p += __shfl_xor(p, 8, 32);
    accp[r] = p;
  }
  if (lanelo == 0) {
#pragma unroll
    for (int r = 0; r < 8; ++r)
      out[base + lanehi * 8 + r] = accp[r] + sScal[0] * sumw + sScal[1];
  }
}

extern "C" void kernel_launch(void* const* d_in, const int* in_sizes, int n_in,
                              void* d_out, int out_size, void* d_ws, size_t ws_size,
                              hipStream_t stream) {
  (void)in_sizes; (void)n_in; (void)d_ws; (void)ws_size; (void)out_size;
  const float* inp  = (const float*)d_in[0];
  const float* Wih0 = (const float*)d_in[1];
  const float* Whh0 = (const float*)d_in[2];
  const float* bih0 = (const float*)d_in[3];
  const float* bhh0 = (const float*)d_in[4];
  const float* Wih1 = (const float*)d_in[5];
  const float* Whh1 = (const float*)d_in[6];
  const float* bih1 = (const float*)d_in[7];
  const float* bhh1 = (const float*)d_in[8];
  const float* fcw  = (const float*)d_in[9];
  const float* fcb  = (const float*)d_in[10];
  const float* fc1w = (const float*)d_in[11];
  const float* fc1b = (const float*)d_in[12];
  float* out = (float*)d_out;

  const int blocks = BATCH / (ROWS * WPB);  // 2048
  lstm2_fc_wmma_kernel<<<blocks, 256, 0, stream>>>(
      inp, Wih0, Whh0, bih0, bhh0, Wih1, Whh1, bih1, bhh1,
      fcw, fcb, fc1w, fc1b, out);
}
